// ThreeLayerGIN_29094108463692
// MI455X (gfx1250) — compile-verified, matrix operand order checked by reading, not run on hardware
//
#include <hip/hip_runtime.h>
#include <stdint.h>

typedef __attribute__((ext_vector_type(16))) __bf16 v16bf;
typedef __attribute__((ext_vector_type(8)))  float  v8f;
typedef int gin_v4i __attribute__((vector_size(16)));

union BFrag { v16bf v; uint4 q[2]; };

#if __has_builtin(__builtin_amdgcn_global_load_async_to_lds_b128)
#define GIN_ASYNC_LDS 1
#endif

__device__ __forceinline__ unsigned short f2bf(float x) {
  union { float f; unsigned u; } c; c.f = x;
  unsigned r = c.u + 0x7FFFu + ((c.u >> 16) & 1u);   // round-to-nearest-even
  return (unsigned short)(r >> 16);
}

// ---------------- elementwise / utility kernels ----------------

__global__ void k_zero(float* __restrict__ p, long n) {
  long i = (long)blockIdx.x * blockDim.x + threadIdx.x;
  long s = (long)gridDim.x * blockDim.x;
  for (; i < n; i += s) p[i] = 0.0f;
}

// f32 [K,N] row-major -> bf16 [N,K] (transposed, N-major)
__global__ void k_wtr(const float* __restrict__ w, unsigned short* __restrict__ wt,
                      int K, int Nw) {
  int i = blockIdx.x * blockDim.x + threadIdx.x;
  if (i >= K * Nw) return;
  int k = i / Nw, n = i - k * Nw;
  wt[n * K + k] = f2bf(w[i]);
}

// agg[dst] += h[src]   (one thread per edge x float4)
__global__ void k_scatter(const float* __restrict__ h, const long long* __restrict__ src,
                          const long long* __restrict__ dst, float* __restrict__ agg,
                          int E, int D) {
  int qpe = D >> 2;
  long total = (long)E * qpe;
  long tid = (long)blockIdx.x * blockDim.x + threadIdx.x;
  if (tid >= total) return;
  int e = (int)(tid / qpe);
  int q = (int)(tid - (long)e * qpe);
  int s = (int)src[e];
  int d = (int)dst[e];
  float4 v = *(const float4*)(h + (long)s * D + (q << 2));
  float* ap = agg + (long)d * D + (q << 2);
  atomicAdd(ap + 0, v.x);
  atomicAdd(ap + 1, v.y);
  atomicAdd(ap + 2, v.z);
  atomicAdd(ap + 3, v.w);
}

// z = bf16(h + agg)
__global__ void k_prep(const float* __restrict__ h, const float* __restrict__ agg,
                       unsigned short* __restrict__ z, long n) {
  long i = (long)blockIdx.x * blockDim.x + threadIdx.x;
  long s = (long)gridDim.x * blockDim.x;
  for (; i < n; i += s) z[i] = f2bf(h[i] + agg[i]);
}

// column sums / sums-of-squares over 256 channels (thread == channel, coalesced rows)
__global__ void k_bnstats(const float* __restrict__ y, float* __restrict__ stats, int rows) {
  int c = threadIdx.x;
  int r0 = blockIdx.x * 128;
  int r1 = r0 + 128 < rows ? r0 + 128 : rows;
  float s = 0.0f, s2 = 0.0f;
  for (int r = r0; r < r1; ++r) {
    float v = y[(long)r * 256 + c];
    s += v; s2 += v * v;
  }
  atomicAdd(&stats[c], s);
  atomicAdd(&stats[256 + c], s2);
}

__global__ void k_bnrelu_bf(const float* __restrict__ y, const float* __restrict__ stats,
                            const float* __restrict__ g, const float* __restrict__ be,
                            unsigned short* __restrict__ out, long n, float invN) {
  long i = (long)blockIdx.x * blockDim.x + threadIdx.x;
  long s = (long)gridDim.x * blockDim.x;
  for (; i < n; i += s) {
    int c = (int)(i & 255);
    float mu  = stats[c] * invN;
    float var = stats[256 + c] * invN - mu * mu;
    float r = (y[i] - mu) * rsqrtf(var + 1e-5f) * g[c] + be[c];
    out[i] = f2bf(r > 0.0f ? r : 0.0f);
  }
}

__global__ void k_bnrelu_f(const float* __restrict__ y, const float* __restrict__ stats,
                           const float* __restrict__ g, const float* __restrict__ be,
                           float* __restrict__ out, long n, float invN) {
  long i = (long)blockIdx.x * blockDim.x + threadIdx.x;
  long s = (long)gridDim.x * blockDim.x;
  for (; i < n; i += s) {
    int c = (int)(i & 255);
    float mu  = stats[c] * invN;
    float var = stats[256 + c] * invN - mu * mu;
    float r = (y[i] - mu) * rsqrtf(var + 1e-5f) * g[c] + be[c];
    out[i] = r > 0.0f ? r : 0.0f;
  }
}

// ---------------- WMMA GEMM with LDS-staged B panels ----------------
// out[M, NOUT] = A[M,K](bf16) * W[K, NOUT] + bias.  WT = W transposed [NOUT,K] bf16.
// Block: (NOUT/128)*4 waves. Wave w: M-strip (w&3)*16 rows, N-half (w>>2)*128 cols.
// Each wave: 8 accumulator tiles (64 VGPRs). K processed in 64-wide panels staged
// in LDS via async copy; rows padded to 72 elems for conflict-free b128 reads.
// B fragments pipelined in pairs with one-pair lookahead; both A fragments of a
// panel preloaded so global latency overlaps the first WMMAs.
template <int NOUT>
__global__ __launch_bounds__(256)
void k_gemm_lds(const unsigned short* __restrict__ A,
                const unsigned short* __restrict__ WT,
                const float* __restrict__ bias, float* __restrict__ out,
                int M, int K) {
  constexpr int LDSROW = 72;                       // 64 elems + 8 pad
  __shared__ unsigned short sB[NOUT * LDSROW];     // <= 36 KB

  const int lane  = threadIdx.x & 31;
  const int w     = threadIdx.x >> 5;
  const int laneM = lane & 15;
  const int cb    = (lane >> 4) << 3;              // K-chunk base per ISA A/B layout
  const int nBase = (w >> 2) << 7;                 // N-half base (0 or 128)

  int row  = blockIdx.x * 64 + (w & 3) * 16 + laneM;
  int rowc = row < M ? row : M - 1;                // clamp; stores are guarded
  const unsigned short* arow = A + (long)rowc * K;

  v8f acc[8];
#pragma unroll
  for (int t = 0; t < 8; ++t) acc[t] = v8f{};

  auto loadB = [&](BFrag& b, int t, int kk) {
    const unsigned short* bp = sB + (nBase + t * 16 + laneM) * LDSROW + kk + cb;
    b.q[0] = *(const uint4*)bp;
    b.q[1] = *(const uint4*)(bp + 16);
  };

  const int chunks = NOUT * 8;                     // 16B chunks per 64-K panel
  for (int k0 = 0; k0 < K; k0 += 64) {
    __syncthreads();                               // previous panel fully consumed
    for (int c = threadIdx.x; c < chunks; c += (NOUT / 128) * 128) {
      int n   = c >> 3;
      int col = (c & 7) << 3;                      // element offset in panel row
      const unsigned short* g = WT + (long)n * K + k0 + col;
      unsigned short*       l = sB + n * LDSROW + col;
#ifdef GIN_ASYNC_LDS
      __builtin_amdgcn_global_load_async_to_lds_b128(
          (__attribute__((address_space(1))) gin_v4i*)g,
          (__attribute__((address_space(3))) gin_v4i*)l, 0, 0);
#else
      *(uint4*)l = *(const uint4*)g;
#endif
    }
#ifdef GIN_ASYNC_LDS
    asm volatile("s_wait_asynccnt 0x0" ::: "memory");
#endif
    __syncthreads();

    // preload both A fragments of this panel (global latency overlaps WMMAs)
    BFrag a[2];
#pragma unroll
    for (int j = 0; j < 2; ++j) {
      a[j].q[0] = *(const uint4*)(arow + k0 + j * 32 + cb);
      a[j].q[1] = *(const uint4*)(arow + k0 + j * 32 + cb + 16);
    }

#pragma unroll
    for (int j = 0; j < 2; ++j) {
      const int kk = j * 32;
      BFrag c0, c1, n0, n1;
      loadB(c0, 0, kk);
      loadB(c1, 1, kk);
#pragma unroll
      for (int t = 0; t < 8; t += 2) {
        if (t + 2 < 8) {                           // one-pair lookahead
          loadB(n0, t + 2, kk);
          loadB(n1, t + 3, kk);
        }
        acc[t]     = __builtin_amdgcn_wmma_f32_16x16x32_bf16(
            false, a[j].v, false, c0.v, (short)0, acc[t], false, false);
        acc[t + 1] = __builtin_amdgcn_wmma_f32_16x16x32_bf16(
            false, a[j].v, false, c1.v, (short)0, acc[t + 1], false, false);
        c0 = n0;
        c1 = n1;
      }
    }
  }

  // D layout: VGPR v -> m = v + (lane>=16 ? 8 : 0); n = lane%16
  const int mBase = blockIdx.x * 64 + (w & 3) * 16 + cb;
#pragma unroll
  for (int t = 0; t < 8; ++t) {
    int n = nBase + t * 16 + laneM;
    float bv = bias[n];
#pragma unroll
    for (int v = 0; v < 8; ++v) {
      int m = mBase + v;
      if (m < M) out[(long)m * NOUT + n] = acc[t][v] + bv;
    }
  }
}

// ---------------- host ----------------

extern "C" void kernel_launch(void* const* d_in, const int* in_sizes, int n_in,
                              void* d_out, int out_size, void* d_ws, size_t ws_size,
                              hipStream_t stream) {
  (void)n_in; (void)out_size; (void)ws_size;
  const float*     x   = (const float*)d_in[0];
  const long long* ei  = (const long long*)d_in[1];
  const float* w1a = (const float*)d_in[2],  *b1a = (const float*)d_in[3];
  const float* g1a = (const float*)d_in[4],  *be1a= (const float*)d_in[5];
  const float* w1b = (const float*)d_in[6],  *b1b = (const float*)d_in[7];
  const float* g1  = (const float*)d_in[8],  *be1 = (const float*)d_in[9];
  const float* w2a = (const float*)d_in[10], *b2a = (const float*)d_in[11];
  const float* g2a = (const float*)d_in[12], *be2a= (const float*)d_in[13];
  const float* w2b = (const float*)d_in[14], *b2b = (const float*)d_in[15];
  const float* g2  = (const float*)d_in[16], *be2 = (const float*)d_in[17];
  const float* w3a = (const float*)d_in[18], *b3a = (const float*)d_in[19];
  const float* g3a = (const float*)d_in[20], *be3a= (const float*)d_in[21];
  const float* w3b = (const float*)d_in[22], *b3b = (const float*)d_in[23];

  const int N = in_sizes[0] / 128;
  const int E = in_sizes[1] / 2;
  const long long* src  = ei;
  const long long* dstp = ei + E;

  uint8_t* p = (uint8_t*)d_ws;
  auto carve = [&](size_t bytes) -> void* {
    void* r = (void*)p; p += (bytes + 255) & ~(size_t)255; return r;
  };
  float*          hbuf = (float*)carve((size_t)N * 256 * 4);
  float*          agg  = (float*)carve((size_t)N * 256 * 4);
  float*          y1   = agg;                       // agg consumed before GEMM1 writes
  unsigned short* zbf  = (unsigned short*)carve((size_t)N * 256 * 2);
  unsigned short* abf  = (unsigned short*)carve((size_t)N * 256 * 2);
  float*          y2   = (float*)carve((size_t)N * 256 * 4);
  float*          stats= (float*)carve(512 * 4);
  unsigned short* w1aT = (unsigned short*)carve(256 * 128 * 2);
  unsigned short* w1bT = (unsigned short*)carve(256 * 256 * 2);
  unsigned short* w2aT = (unsigned short*)carve(256 * 256 * 2);
  unsigned short* w2bT = (unsigned short*)carve(256 * 256 * 2);
  unsigned short* w3aT = (unsigned short*)carve(256 * 256 * 2);
  unsigned short* w3bT = (unsigned short*)carve(128 * 256 * 2);

  // weight convert+transpose (tiny)
  k_wtr<<<(128 * 256 + 255) / 256, 256, 0, stream>>>(w1a, w1aT, 128, 256);
  k_wtr<<<(256 * 256 + 255) / 256, 256, 0, stream>>>(w1b, w1bT, 256, 256);
  k_wtr<<<(256 * 256 + 255) / 256, 256, 0, stream>>>(w2a, w2aT, 256, 256);
  k_wtr<<<(256 * 256 + 255) / 256, 256, 0, stream>>>(w2b, w2bT, 256, 256);
  k_wtr<<<(256 * 256 + 255) / 256, 256, 0, stream>>>(w3a, w3aT, 256, 256);
  k_wtr<<<(256 * 128 + 255) / 256, 256, 0, stream>>>(w3b, w3bT, 256, 128);

  const float invN = 1.0f / (float)N;
  const int gemmGrid = (N + 63) / 64;
  const int statGrid = (N + 127) / 128;

  auto gin = [&](const float* h, int D,
                 const unsigned short* WaT, const float* ba,
                 const float* ga, const float* bea,
                 const unsigned short* WbT, const float* bb,
                 int Nout, float* out2) {
    long nD = (long)N * D;
    k_zero<<<1024, 256, 0, stream>>>(agg, nD);
    long st = (long)E * (D / 4);
    k_scatter<<<(int)((st + 255) / 256), 256, 0, stream>>>(h, src, dstp, agg, E, D);
    k_prep<<<1024, 256, 0, stream>>>(h, agg, zbf, nD);
    k_gemm_lds<256><<<gemmGrid, 256, 0, stream>>>(zbf, WaT, ba, y1, N, D);
    k_zero<<<1, 512, 0, stream>>>(stats, 512);
    k_bnstats<<<statGrid, 256, 0, stream>>>(y1, stats, N);
    k_bnrelu_bf<<<1024, 256, 0, stream>>>(y1, stats, ga, bea, abf, (long)N * 256, invN);
    if (Nout == 256)
      k_gemm_lds<256><<<gemmGrid, 256, 0, stream>>>(abf, WbT, bb, out2, N, 256);
    else
      k_gemm_lds<128><<<gemmGrid, 128, 0, stream>>>(abf, WbT, bb, out2, N, 256);
  };

  // layer 1
  gin(x, 128, w1aT, b1a, g1a, be1a, w1bT, b1b, 256, y2);
  k_zero<<<1, 512, 0, stream>>>(stats, 512);
  k_bnstats<<<statGrid, 256, 0, stream>>>(y2, stats, N);
  k_bnrelu_f<<<1024, 256, 0, stream>>>(y2, stats, g1, be1, hbuf, (long)N * 256, invN);

  // layer 2
  gin(hbuf, 256, w2aT, b2a, g2a, be2a, w2bT, b2b, 256, y2);
  k_zero<<<1, 512, 0, stream>>>(stats, 512);
  k_bnstats<<<statGrid, 256, 0, stream>>>(y2, stats, N);
  k_bnrelu_f<<<1024, 256, 0, stream>>>(y2, stats, g2, be2, hbuf, (long)N * 256, invN);

  // layer 3 (no outer BN) -> d_out
  gin(hbuf, 256, w3aT, b3a, g3a, be3a, w3bT, b3b, 128, (float*)d_out);
}
